// MetaEncoder_77799037599906
// MI455X (gfx1250) — matrix operand order repaired and checked by reading, not compiled, not run on hardware
//
#include <hip/hip_runtime.h>
#include <hip/hip_bf16.h>

// ---------------------------------------------------------------------------
// 2-layer GCN for MI455X (gfx1250, wave32, WMMA)
//   layer: h = X @ W^T ; agg = scatter_add(h[src]*dinv[src]*dinv[dst] -> dst)
//                        + h*dinv^2 + b ; (relu after layer 1)
// GEMMs: bf16 WMMA (v_wmma_f32_16x16x32_bf16) with 32Mx64N register blocking
// per wave (8 accumulators, 1.5 b128-loads per WMMA) and a fused
// self-loop+bias epilogue. Aggregation: L2-resident gather/atomic-scatter
// (h and agg each ~102MB < 192MB L2).
// ---------------------------------------------------------------------------

typedef __attribute__((ext_vector_type(16))) __bf16 v16bf;
typedef __attribute__((ext_vector_type(8)))  float  v8f;

struct B32x8 { unsigned int w[8]; };

__device__ __forceinline__ unsigned short f2bf_rne(float f) {
    unsigned int u = __float_as_uint(f);
    unsigned int r = u + 0x7FFFu + ((u >> 16) & 1u);   // round-to-nearest-even
    return (unsigned short)(r >> 16);
}

// ---- elementwise converts ------------------------------------------------

__global__ void k_f32_to_bf16(const float* __restrict__ src,
                              unsigned short* __restrict__ dst, long long n) {
    long long i = (long long)blockIdx.x * blockDim.x + threadIdx.x;
    if (i < n) dst[i] = f2bf_rne(src[i]);
}

__global__ void k_relu_f32_to_bf16(const float* __restrict__ src,
                                   unsigned short* __restrict__ dst, long long n) {
    long long i = (long long)blockIdx.x * blockDim.x + threadIdx.x;
    if (i < n) dst[i] = f2bf_rne(fmaxf(src[i], 0.0f));
}

// ---- degree / rsqrt normalization ---------------------------------------

__global__ void k_deg_init(float* __restrict__ dinv, int n) {
    int i = blockIdx.x * blockDim.x + threadIdx.x;
    if (i < n) dinv[i] = 1.0f;                          // +1 self loop
}

__global__ void k_deg_acc(const int* __restrict__ ei, float* __restrict__ dinv, int E) {
    int e = blockIdx.x * blockDim.x + threadIdx.x;
    if (e < E) atomicAdd(&dinv[ei[E + e]], 1.0f);       // dst row of edge_index
}

__global__ void k_deg_rsqrt(float* __restrict__ dinv, int n) {
    int i = blockIdx.x * blockDim.x + threadIdx.x;
    if (i < n) dinv[i] = rsqrtf(dinv[i]);
}

// ---- WMMA bf16 GEMM + fused self-loop/bias epilogue ----------------------
// H[M x Cout] = A[M x K](bf16) * W[Cout x K](bf16)^T
// AGG[m,n]    = H[m,n] * dinv[m]^2 + bias[n]
// Wave tile: 32M x 64N  (2 A-frags x 4 B-frags -> 8 accumulators)
// Block: 8 waves -> 32M x 512N panel (layer2 Cout=256 masks upper waves).
// Frag layouts per ISA tables:
//   A: lane&15 = M row, half=lane>>4 -> K in {h*8..h*8+7, 16+h*8..16+h*8+7}
//   B: lane&15 = N col (row of W),    half -> K in {h*8..h*8+7, 16+h*8..16+h*8+7}
//   D: element v -> M = v + half*8, N = lane&15

__global__ __launch_bounds__(256)
void k_wmma_gemm_bf16(const unsigned short* __restrict__ A,
                      const unsigned short* __restrict__ W,
                      float* __restrict__ H, float* __restrict__ AGG,
                      const float* __restrict__ dinv, const float* __restrict__ bias,
                      int M, int K, int Cout) {
    const int lane = threadIdx.x & 31;
    const int wave = threadIdx.x >> 5;
    const int idx  = lane & 15;
    const int half = lane >> 4;

    const int m0 = blockIdx.x * 32;
    const int n0 = wave * 64;
    if (n0 >= Cout) return;

    // A row pointers (2 M sub-tiles); clamp reads for the possibly-OOB tail tile
    const unsigned short* arow[2];
#pragma unroll
    for (int s = 0; s < 2; ++s) {
        int r = m0 + s * 16 + idx;
        if (r > M - 1) r = M - 1;
        arow[s] = A + (size_t)r * K;
    }
    const unsigned short* brow[4];
#pragma unroll
    for (int t = 0; t < 4; ++t)
        brow[t] = W + (size_t)(n0 + t * 16 + idx) * K;

    v8f acc[2][4] = {};
    for (int k0 = 0; k0 < K; k0 += 32) {
        v16bf af[2], bf[4];
        const int ha = half * 4;                        // half*8 bf16 = 4 dwords
        const int hb = half * 4;                        // half*8 bf16 = 4 dwords
#pragma unroll
        for (int s = 0; s < 2; ++s) {
            const unsigned int* ap = (const unsigned int*)(arow[s] + k0);
            B32x8 w;
#pragma unroll
            for (int j = 0; j < 4; ++j) { w.w[j] = ap[ha + j]; w.w[4 + j] = ap[8 + ha + j]; }
            af[s] = __builtin_bit_cast(v16bf, w);
        }
#pragma unroll
        for (int t = 0; t < 4; ++t) {
            const unsigned int* bp = (const unsigned int*)(brow[t] + k0);
            B32x8 w;
#pragma unroll
            for (int j = 0; j < 4; ++j) { w.w[j] = bp[hb + j]; w.w[4 + j] = bp[8 + hb + j]; }
            bf[t] = __builtin_bit_cast(v16bf, w);
        }
        if (k0 + 32 < K) {                              // gfx1250 global_prefetch
            __builtin_prefetch(arow[0] + k0 + 32, 0, 3);
            __builtin_prefetch(arow[1] + k0 + 32, 0, 3);
            __builtin_prefetch(brow[0] + k0 + 32, 0, 3);
            __builtin_prefetch(brow[2] + k0 + 32, 0, 3);
        }
#pragma unroll
        for (int s = 0; s < 2; ++s)
#pragma unroll
            for (int t = 0; t < 4; ++t)
                acc[s][t] = __builtin_amdgcn_wmma_f32_16x16x32_bf16(
                    /*neg_a=*/false, af[s], /*neg_b=*/false, bf[t],
                    /*c_mod=*/(short)0, acc[s][t],
                    /*reuse_a=*/false, /*reuse_b=*/false);
    }

    // epilogue: H = gemm result ; AGG = H*dinv^2 + bias  (fused self-loop init)
#pragma unroll
    for (int s = 0; s < 2; ++s) {
        const int mt = m0 + s * 16;
        if (mt < M) {
#pragma unroll
            for (int v = 0; v < 8; ++v) {
                const int m = mt + half * 8 + v;
                float dv = dinv[m]; dv *= dv;
                float* hrow = H   + (size_t)m * Cout;
                float* grow = AGG + (size_t)m * Cout;
#pragma unroll
                for (int t = 0; t < 4; ++t) {
                    const int n = n0 + t * 16 + idx;
                    const float val = acc[s][t][v];
                    hrow[n] = val;
                    grow[n] = fmaf(val, dv, bias[n]);
                }
            }
        }
    }
}

// ---- edge scatter: out[dst,c] += h[src,c] * dinv[src]*dinv[dst] ----------
// Consecutive lanes = consecutive channels of one edge (coalesced gather,
// 4x global_atomic_add_f32 per lane, L2-resident working set).

__global__ void k_edge_scatter(const float* __restrict__ h, const int* __restrict__ ei,
                               const float* __restrict__ dinv, float* __restrict__ out,
                               int E, int C) {
    const int cb4 = C >> 2;
    long long i = (long long)blockIdx.x * blockDim.x + threadIdx.x;
    long long total = (long long)E * cb4;
    if (i >= total) return;
    int e = (int)(i / cb4);
    int c = (int)(i % cb4) * 4;
    int s = ei[e];
    int d = ei[E + e];
    float norm = dinv[s] * dinv[d];
    float4 hv = *(const float4*)(h + (size_t)s * C + c);
    float* op = out + (size_t)d * C + c;
    atomicAdd(op + 0, hv.x * norm);
    atomicAdd(op + 1, hv.y * norm);
    atomicAdd(op + 2, hv.z * norm);
    atomicAdd(op + 3, hv.w * norm);
}

// ---------------------------------------------------------------------------

static inline size_t align256(size_t v) { return (v + 255) & ~(size_t)255; }

extern "C" void kernel_launch(void* const* d_in, const int* in_sizes, int n_in,
                              void* d_out, int out_size, void* d_ws, size_t ws_size,
                              hipStream_t stream) {
    const float* x  = (const float*)d_in[0];
    const int*   ei = (const int*)d_in[1];
    const float* w1 = (const float*)d_in[2];
    const float* b1 = (const float*)d_in[3];
    const float* w2 = (const float*)d_in[4];
    const float* b2 = (const float*)d_in[5];
    float* out = (float*)d_out;

    const int CIN = 512, C1 = 512, C2 = 256;
    const int N = in_sizes[0] / CIN;        // 50000 (multiple of 16, not 32)
    const int E = in_sizes[1] / 2;          // 1.6M

    // workspace layout
    char* ws = (char*)d_ws;
    size_t off = 0;
    float* dinv = (float*)(ws + off);                  off += align256((size_t)N * 4);
    unsigned short* w1b = (unsigned short*)(ws + off); off += align256((size_t)C1 * CIN * 2);
    unsigned short* w2b = (unsigned short*)(ws + off); off += align256((size_t)C2 * CIN * 2);
    unsigned short* xb  = (unsigned short*)(ws + off); off += align256((size_t)N * CIN * 2); // x bf16, reused as h1 bf16
    float* hbuf = (float*)(ws + off);                  off += align256((size_t)N * C1 * 4);  // raw gemm out (L1 then L2)
    float* agg1 = (float*)(ws + off);                  off += align256((size_t)N * C1 * 4);

    const int TB = 256;
    auto grid1d = [](long long n, int tb) { return (unsigned)((n + tb - 1) / tb); };
    const unsigned mblocks = (unsigned)((N + 31) / 32);

    // 0) fp32 -> bf16 operands
    k_f32_to_bf16<<<grid1d((long long)N * CIN, TB), TB, 0, stream>>>(x, xb, (long long)N * CIN);
    k_f32_to_bf16<<<grid1d((long long)C1 * CIN, TB), TB, 0, stream>>>(w1, w1b, (long long)C1 * CIN);
    k_f32_to_bf16<<<grid1d((long long)C2 * CIN, TB), TB, 0, stream>>>(w2, w2b, (long long)C2 * CIN);

    // 1) degree -> dinv = rsqrt(1 + indeg)
    k_deg_init<<<grid1d(N, TB), TB, 0, stream>>>(dinv, N);
    k_deg_acc<<<grid1d(E, TB), TB, 0, stream>>>(ei, dinv, E);
    k_deg_rsqrt<<<grid1d(N, TB), TB, 0, stream>>>(dinv, N);

    // 2) layer 1: GEMM (+fused self-loop/bias into agg1), edge scatter, relu->bf16
    k_wmma_gemm_bf16<<<mblocks, TB, 0, stream>>>(xb, w1b, hbuf, agg1, dinv, b1, N, CIN, C1);
    k_edge_scatter<<<grid1d((long long)E * (C1 / 4), TB), TB, 0, stream>>>(hbuf, ei, dinv, agg1, E, C1);
    k_relu_f32_to_bf16<<<grid1d((long long)N * C1, TB), TB, 0, stream>>>(agg1, xb, (long long)N * C1);

    // 3) layer 2: GEMM (+fused epilogue straight into d_out), edge scatter
    k_wmma_gemm_bf16<<<mblocks, TB, 0, stream>>>(xb, w2b, hbuf, out, dinv, b2, N, C1, C2);
    k_edge_scatter<<<grid1d((long long)E * (C2 / 4), TB), TB, 0, stream>>>(hbuf, ei, dinv, out, E, C2);
}
